// SelfAttention_47115791237647
// MI455X (gfx1250) — matrix operand order repaired and checked by reading, not compiled, not run on hardware
//
#include <hip/hip_runtime.h>
#include <hip/hip_bf16.h>

// ---------------- problem constants ----------------
#define BATCH    32
#define NTOK     577
#define EMB      768
#define NHEADS   12
#define HD       64
#define NPAD     608            // token padding for attention: 19*32
#define NROWS    (BATCH*NTOK)   // 18464 (divisible by 32)
#define MTILES2  (NROWS/32)     // 577 (32-row M blocks)
#define HGT      24
#define WID      24
#define RELW     (2*WID-1)      // 47
#define KTILES   (EMB/32)       // 24
#define QTILES   37             // ceil(577/16)
#define KBLOCKS  (NPAD/32)      // 19

typedef __attribute__((ext_vector_type(16))) _Float16 v16h;
typedef __attribute__((ext_vector_type(4)))  _Float16 v4h;
typedef __attribute__((ext_vector_type(8)))  float    v8f;
typedef __attribute__((ext_vector_type(4)))  float    f32x4;

union ABFrag {        // 32 bytes: one 16x32 (or 32x16) f16 WMMA operand per lane
    v16h  h;
    f32x4 q[2];
};

static __device__ __forceinline__ v8f wmma_f16(const v16h& a, const v16h& b, const v8f& c) {
    return __builtin_amdgcn_wmma_f32_16x16x32_f16(false, a, false, b, (short)0, c, false, false);
}

// ---------------- stage 1: f32 -> f16 convert (vectorized x4) ----------------
__global__ void cvt_f32_f16_x4(const float* __restrict__ s, _Float16* __restrict__ d, int n4) {
    int i = blockIdx.x * blockDim.x + threadIdx.x;
    if (i < n4) {
        f32x4 v = ((const f32x4*)s)[i];
        v4h h = { (_Float16)v[0], (_Float16)v[1], (_Float16)v[2], (_Float16)v[3] };
        ((v4h*)d)[i] = h;
    }
}

// ---------------- stage 2: QKV projection GEMM ----------------
// grid (MTILES2, 3, 3), block 128 (4 waves). Wave computes 32(M) x 64(N) tile
// (2 M-subtiles x 4 N-subtiles register blocking; B fragment reused 2x).
// Q,K -> [b,h,NPAD,HD] (d contiguous); V -> transposed [b,h,HD,NPAD] (tok contiguous).
__global__ __launch_bounds__(128)
void qkv_gemm(const _Float16* __restrict__ xh,
              const _Float16* __restrict__ wq, const _Float16* __restrict__ wk,
              const _Float16* __restrict__ wv,
              const float* __restrict__ bq, const float* __restrict__ bk,
              const float* __restrict__ bv,
              _Float16* __restrict__ qh, _Float16* __restrict__ kh,
              _Float16* __restrict__ vt)
{
    const int mtile = blockIdx.x;                 // 32-row block
    const int wave  = threadIdx.x >> 5;
    const int lane  = threadIdx.x & 31;
    const int mat   = blockIdx.z;                 // 0=Q 1=K 2=V
    const int n0    = (blockIdx.y * 4 + wave) * 64;

    const _Float16* W    = (mat == 0) ? wq : (mat == 1) ? wk : wv;
    const float*    bias = (mat == 0) ? bq : (mat == 1) ? bk : bv;

    const int arow  = mtile * 32 + (lane & 15);
    const int aKoff = (lane & 16) ? 8 : 0;        // A-layout K split per lane half
    const int bKoff = (lane & 16) ? 8 : 0;        // B-layout K split per lane half

    v8f acc[2][4] = {};
    for (int kt = 0; kt < KTILES; ++kt) {
        ABFrag a0, a1;
        {
            const _Float16* ap = xh + (size_t)arow * EMB + kt * 32 + aKoff;
            a0.q[0] = *(const f32x4*)ap;
            a0.q[1] = *(const f32x4*)(ap + 16);
            ap += (size_t)16 * EMB;
            a1.q[0] = *(const f32x4*)ap;
            a1.q[1] = *(const f32x4*)(ap + 16);
        }
#pragma unroll
        for (int nt = 0; nt < 4; ++nt) {
            ABFrag b;
            const _Float16* bp = W + (size_t)(n0 + nt * 16 + (lane & 15)) * EMB + kt * 32 + bKoff;
            b.q[0] = *(const f32x4*)bp;
            b.q[1] = *(const f32x4*)(bp + 16);
            acc[0][nt] = wmma_f16(a0.h, b.h, acc[0][nt]);
            acc[1][nt] = wmma_f16(a1.h, b.h, acc[1][nt]);
        }
    }

#pragma unroll
    for (int mi = 0; mi < 2; ++mi) {
        const int rowBase = mtile * 32 + mi * 16 + ((lane & 16) ? 8 : 0);
#pragma unroll
        for (int nt = 0; nt < 4; ++nt) {
            const int n  = n0 + nt * 16 + (lane & 15);
            const int h  = n >> 6;
            const int d  = n & 63;
            const float bb = bias[n];
#pragma unroll
            for (int r = 0; r < 8; ++r) {
                const int m   = rowBase + r;
                const int b_  = m / NTOK;
                const int tok = m - b_ * NTOK;
                const _Float16 hv = (_Float16)(acc[mi][nt][r] + bb);
                if (mat == 2) {
                    vt[(((size_t)b_ * NHEADS + h) * HD + d) * NPAD + tok] = hv;
                } else {
                    _Float16* dst = (mat == 0) ? qh : kh;
                    dst[(((size_t)b_ * NHEADS + h) * NPAD + tok) * HD + d] = hv;
                }
            }
        }
    }
}

// ---------------- stage 3: fused attention (flash-style) ----------------
// grid (BATCH, NHEADS, QTILES), block 32 (one wave). 16 queries x all keys.
__global__ __launch_bounds__(32)
void attention(const _Float16* __restrict__ qh, const _Float16* __restrict__ kh,
               const _Float16* __restrict__ vt, const float* __restrict__ rel_table,
               _Float16* __restrict__ attn_out /* [NROWS, EMB] f16 */)
{
    __shared__ __align__(16) _Float16 pbuf[16 * 32];

    const int b    = blockIdx.x;
    const int h    = blockIdx.y;
    const int qt   = blockIdx.z;
    const int lane = threadIdx.x;
    const float scale = 0.125f;                  // 1/sqrt(64)

    const int aKoff = (lane & 16) ? 8 : 0;
    const int bKoff = (lane & 16) ? 8 : 0;

    // Q tile (16 x 64) as two A fragments (K-steps of 32)
    const _Float16* Qbase = qh + (((size_t)b * NHEADS + h) * NPAD + qt * 16) * HD;
    ABFrag qa[2];
#pragma unroll
    for (int s = 0; s < 2; ++s) {
        const _Float16* p = Qbase + (size_t)(lane & 15) * HD + s * 32 + aKoff;
        qa[s].q[0] = *(const f32x4*)p;
        qa[s].q[1] = *(const f32x4*)(p + 16);
    }

    // per-lane row metadata (rows r / r+8 by lane half — same mapping for S and O)
    int qok[8], qy[8], qx[8];
#pragma unroll
    for (int r = 0; r < 8; ++r) {
        const int qtok = qt * 16 + ((lane & 16) ? 8 : 0) + r;
        qok[r] = (qtok >= 1 && qtok < NTOK);
        const int t = (qtok >= 1) ? (qtok - 1) : 0;
        qy[r] = t / WID;
        qx[r] = t - qy[r] * WID;
    }

    float mrow[8], lrow[8];
#pragma unroll
    for (int r = 0; r < 8; ++r) { mrow[r] = -1.0e30f; lrow[r] = 0.f; }
    v8f o[4] = {};

    const _Float16* Kbase = kh + ((size_t)b * NHEADS + h) * NPAD * HD;
    const _Float16* Vbase = vt + ((size_t)b * NHEADS + h) * HD * NPAD;

    for (int kb = 0; kb < KBLOCKS; ++kb) {
        // ---- S = (Q Kt) : 2 key sub-tiles x 2 K-steps ----
        v8f s[2] = {};
#pragma unroll
        for (int sub = 0; sub < 2; ++sub) {
            const int kkey = kb * 32 + sub * 16 + (lane & 15);
#pragma unroll
            for (int ks = 0; ks < 2; ++ks) {
                ABFrag kf;
                const _Float16* p = Kbase + (size_t)kkey * HD + ks * 32 + bKoff;
                kf.q[0] = *(const f32x4*)p;
                kf.q[1] = *(const f32x4*)(p + 16);
                s[sub] = wmma_f16(qa[ks].h, kf.h, s[sub]);
            }
        }

        // ---- scale + relative-position bias + padding mask; block row max ----
        float bmax[8];
#pragma unroll
        for (int r = 0; r < 8; ++r) bmax[r] = -1.0e30f;
#pragma unroll
        for (int sub = 0; sub < 2; ++sub) {
            const int  kkey   = kb * 32 + sub * 16 + (lane & 15);
            const bool kvalid = (kkey < NTOK);
            const int  kt_    = (kkey >= 1) ? (kkey - 1) : 0;
            const int  ky     = kt_ / WID;
            const int  kx     = kt_ - ky * WID;
            const bool kbias  = kvalid && (kkey >= 1);
#pragma unroll
            for (int r = 0; r < 8; ++r) {
                float val = s[sub][r] * scale;
                if (kbias && qok[r]) {
                    const int idx = (qy[r] - ky + HGT - 1) * RELW + (qx[r] - kx + WID - 1);
                    val += rel_table[idx * NHEADS + h];
                }
                if (!kvalid) val = -1.0e30f;
                s[sub][r] = val;
                bmax[r] = fmaxf(bmax[r], val);
            }
        }
#pragma unroll
        for (int mk = 1; mk < 16; mk <<= 1)
#pragma unroll
            for (int r = 0; r < 8; ++r)
                bmax[r] = fmaxf(bmax[r], __shfl_xor(bmax[r], mk, 32));

        // ---- online softmax update ----
        float alpha[8];
#pragma unroll
        for (int r = 0; r < 8; ++r) {
            const float mnew = fmaxf(mrow[r], bmax[r]);
            alpha[r] = __expf(mrow[r] - mnew);
            mrow[r]  = mnew;
        }
        float bsum[8];
#pragma unroll
        for (int r = 0; r < 8; ++r) bsum[r] = 0.f;
#pragma unroll
        for (int sub = 0; sub < 2; ++sub)
#pragma unroll
            for (int r = 0; r < 8; ++r) {
                const float p = __expf(s[sub][r] - mrow[r]);
                s[sub][r] = p;
                bsum[r] += p;
            }
#pragma unroll
        for (int mk = 1; mk < 16; mk <<= 1)
#pragma unroll
            for (int r = 0; r < 8; ++r)
                bsum[r] += __shfl_xor(bsum[r], mk, 32);
#pragma unroll
        for (int r = 0; r < 8; ++r) lrow[r] = lrow[r] * alpha[r] + bsum[r];
#pragma unroll
        for (int dt = 0; dt < 4; ++dt)
#pragma unroll
            for (int r = 0; r < 8; ++r) o[dt][r] *= alpha[r];

        // ---- repack P: C-layout -> A-layout via LDS ----
#pragma unroll
        for (int sub = 0; sub < 2; ++sub)
#pragma unroll
            for (int r = 0; r < 8; ++r) {
                const int row_ = ((lane & 16) ? 8 : 0) + r;
                pbuf[row_ * 32 + sub * 16 + (lane & 15)] = (_Float16)s[sub][r];
            }
        __syncthreads();
        ABFrag pa;
        {
            const _Float16* pp = pbuf + (lane & 15) * 32 + aKoff;
            pa.q[0] = *(const f32x4*)pp;
            pa.q[1] = *(const f32x4*)(pp + 16);
        }

        // ---- O += P V : 4 d-tiles, V from transposed layout (tok contiguous) ----
#pragma unroll
        for (int dt = 0; dt < 4; ++dt) {
            ABFrag vf;
            const _Float16* p = Vbase + (size_t)(dt * 16 + (lane & 15)) * NPAD + kb * 32 + bKoff;
            vf.q[0] = *(const f32x4*)p;
            vf.q[1] = *(const f32x4*)(p + 16);
            o[dt] = wmma_f16(pa.h, vf.h, o[dt]);
        }
        __syncthreads();
    }

    // ---- normalize + store f16 to [B*NTOK, EMB] ----
#pragma unroll
    for (int dt = 0; dt < 4; ++dt)
#pragma unroll
        for (int r = 0; r < 8; ++r) {
            const int qtok = qt * 16 + ((lane & 16) ? 8 : 0) + r;
            if (qtok < NTOK) {
                const float v = o[dt][r] / lrow[r];
                attn_out[((size_t)(b * NTOK + qtok)) * EMB + h * HD + dt * 16 + (lane & 15)] =
                    (_Float16)v;
            }
        }
}

// ---------------- stage 4: output projection, f32 result ----------------
// grid (MTILES2, 3, 1), block 128. Wave computes 32(M) x 64(N) tile.
__global__ __launch_bounds__(128)
void proj_gemm(const _Float16* __restrict__ ah, const _Float16* __restrict__ wp,
               const float* __restrict__ bp, float* __restrict__ out)
{
    const int mtile = blockIdx.x;
    const int wave  = threadIdx.x >> 5;
    const int lane  = threadIdx.x & 31;
    const int n0    = (blockIdx.y * 4 + wave) * 64;

    const int arow  = mtile * 32 + (lane & 15);
    const int aKoff = (lane & 16) ? 8 : 0;
    const int bKoff = (lane & 16) ? 8 : 0;

    v8f acc[2][4] = {};
    for (int kt = 0; kt < KTILES; ++kt) {
        ABFrag a0, a1;
        {
            const _Float16* ap = ah + (size_t)arow * EMB + kt * 32 + aKoff;
            a0.q[0] = *(const f32x4*)ap;
            a0.q[1] = *(const f32x4*)(ap + 16);
            ap += (size_t)16 * EMB;
            a1.q[0] = *(const f32x4*)ap;
            a1.q[1] = *(const f32x4*)(ap + 16);
        }
#pragma unroll
        for (int nt = 0; nt < 4; ++nt) {
            ABFrag b;
            const _Float16* bpp = wp + (size_t)(n0 + nt * 16 + (lane & 15)) * EMB + kt * 32 + bKoff;
            b.q[0] = *(const f32x4*)bpp;
            b.q[1] = *(const f32x4*)(bpp + 16);
            acc[0][nt] = wmma_f16(a0.h, b.h, acc[0][nt]);
            acc[1][nt] = wmma_f16(a1.h, b.h, acc[1][nt]);
        }
    }

#pragma unroll
    for (int mi = 0; mi < 2; ++mi) {
        const int rowBase = mtile * 32 + mi * 16 + ((lane & 16) ? 8 : 0);
#pragma unroll
        for (int nt = 0; nt < 4; ++nt) {
            const int n = n0 + nt * 16 + (lane & 15);
            const float bb = bp[n];
#pragma unroll
            for (int r = 0; r < 8; ++r)
                out[(size_t)(rowBase + r) * EMB + n] = acc[mi][nt][r] + bb;
        }
    }
}

// ---------------- host ----------------
extern "C" void kernel_launch(void* const* d_in, const int* in_sizes, int n_in,
                              void* d_out, int out_size, void* d_ws, size_t ws_size,
                              hipStream_t stream) {
    const float* x   = (const float*)d_in[0];
    const float* Wq  = (const float*)d_in[1];
    const float* bq  = (const float*)d_in[2];
    const float* Wk  = (const float*)d_in[3];
    const float* bk  = (const float*)d_in[4];
    const float* Wv  = (const float*)d_in[5];
    const float* bv  = (const float*)d_in[6];
    const float* Wp  = (const float*)d_in[7];
    const float* bp  = (const float*)d_in[8];
    const float* rel = (const float*)d_in[9];
    float* out = (float*)d_out;

    char* ws = (char*)d_ws;
    size_t off = 0;
    auto alloc = [&](size_t bytes) -> void* {
        off = (off + 255) & ~(size_t)255;
        void* p = ws + off;
        off += bytes;
        return p;
    };
    _Float16* xh   = (_Float16*)alloc((size_t)NROWS * EMB * 2);
    _Float16* wqh  = (_Float16*)alloc((size_t)EMB * EMB * 2);
    _Float16* wkh  = (_Float16*)alloc((size_t)EMB * EMB * 2);
    _Float16* wvh  = (_Float16*)alloc((size_t)EMB * EMB * 2);
    _Float16* wph  = (_Float16*)alloc((size_t)EMB * EMB * 2);
    _Float16* qhb  = (_Float16*)alloc((size_t)BATCH * NHEADS * NPAD * HD * 2);
    _Float16* khb  = (_Float16*)alloc((size_t)BATCH * NHEADS * NPAD * HD * 2);
    _Float16* vtb  = (_Float16*)alloc((size_t)BATCH * NHEADS * HD * NPAD * 2);
    _Float16* ath  = (_Float16*)alloc((size_t)NROWS * EMB * 2);

    // stage 1: converts
    {
        const int nx = NROWS * EMB / 4;
        cvt_f32_f16_x4<<<(nx + 255) / 256, 256, 0, stream>>>(x, xh, nx);
        const int nw = EMB * EMB / 4;
        cvt_f32_f16_x4<<<(nw + 255) / 256, 256, 0, stream>>>(Wq, wqh, nw);
        cvt_f32_f16_x4<<<(nw + 255) / 256, 256, 0, stream>>>(Wk, wkh, nw);
        cvt_f32_f16_x4<<<(nw + 255) / 256, 256, 0, stream>>>(Wv, wvh, nw);
        cvt_f32_f16_x4<<<(nw + 255) / 256, 256, 0, stream>>>(Wp, wph, nw);
    }
    // stage 2: QKV
    qkv_gemm<<<dim3(MTILES2, 3, 3), 128, 0, stream>>>(xh, wqh, wkh, wvh, bq, bk, bv,
                                                      qhb, khb, vtb);
    // stage 3: fused attention
    attention<<<dim3(BATCH, NHEADS, QTILES), 32, 0, stream>>>(qhb, khb, vtb, rel, ath);
    // stage 4: output projection
    proj_gemm<<<dim3(MTILES2, 3, 1), 128, 0, stream>>>(ath, wph, bp, out);
}
